// SQXNet_10823317585986
// MI455X (gfx1250) — compile-verified, matrix operand order checked by reading, not run on hardware
//
#include <hip/hip_runtime.h>
#include <hip/hip_bf16.h>
#include <stdint.h>

#define BATCH        8192
#define STATE_DIM    512
#define BASE_HIDDEN  2048
#define HEAD_IN      1024
#define HEAD_HIDDEN  1024
#define HEAD_OUT     512
#define NUM_INT      16

typedef __attribute__((ext_vector_type(16))) __bf16        v16bf;
typedef __attribute__((ext_vector_type(8)))  float         v8f;
typedef __attribute__((ext_vector_type(4)))  unsigned int  v4u;

union FragAB { v16bf v; v4u q[2]; };

// f32 -> bf16 round-to-nearest-even
static __device__ __forceinline__ unsigned short f2bf(float f) {
  unsigned int u = __float_as_uint(f);
  u += 0x7FFFu + ((u >> 16) & 1u);
  return (unsigned short)(u >> 16);
}

static __device__ __forceinline__ float elu(float v) {
  return v > 0.0f ? v : __expf(v) - 1.0f;
}

// Load one 16x32 bf16 WMMA operand fragment from a row-major [rows, K] array.
// Per CDNA5 ISA 16-bit A layout: lane (0-15): K = khalf*8+0..7 in VGPR0-3,
// K = 16+khalf*8+0..7 in VGPR4-7.  Same pattern serves B when B is stored
// transposed [N, K] row-major.
static __device__ __forceinline__ v16bf load_frag_g(const unsigned short* row_ptr,
                                                    int k0, int khalf) {
  FragAB f;
  const unsigned short* p = row_ptr + k0 + khalf * 8;
  f.q[0] = *reinterpret_cast<const v4u*>(p);
  f.q[1] = *reinterpret_cast<const v4u*>(p + 16);
  return f.v;
}

static __device__ __forceinline__ v8f wmma_bf16(v16bf a, v16bf b, v8f c) {
  return __builtin_amdgcn_wmma_f32_16x16x32_bf16(false, a, false, b, (short)0,
                                                 c, false, false);
}

// ---------------------------------------------------------------- bucketing
__global__ void bucket_init(int* counts) {
  if (threadIdx.x < NUM_INT) counts[threadIdx.x] = 0;
}

__global__ void bucket(const int* __restrict__ intent, int* counts, int* idx) {
  int b = blockIdx.x * blockDim.x + threadIdx.x;
  if (b < BATCH) {
    int n = intent[b];
    int p = atomicAdd(&counts[n], 1);
    idx[n * BATCH + p] = b;
  }
}

// ----------------------------------------------------------- f32 -> bf16 copy
__global__ void cvt_bf16(const float* __restrict__ in, unsigned short* __restrict__ out,
                         size_t total) {
  for (size_t t = (size_t)blockIdx.x * blockDim.x + threadIdx.x; t < total;
       t += (size_t)gridDim.x * blockDim.x)
    out[t] = f2bf(in[t]);
}

// f32 [K,N] (per batch slice) -> bf16 [N,K]; batch via blockIdx.y
__global__ void transpose_cvt_bf16(const float* __restrict__ in,
                                   unsigned short* __restrict__ out,
                                   int K, int N) {
  size_t total = (size_t)K * N;
  const float* inb = in + (size_t)blockIdx.y * total;
  unsigned short* outb = out + (size_t)blockIdx.y * total;
  for (size_t t = (size_t)blockIdx.x * blockDim.x + threadIdx.x; t < total;
       t += (size_t)gridDim.x * blockDim.x) {
    size_t nn = t / (size_t)K;
    size_t kk = t - nn * (size_t)K;
    outb[t] = f2bf(inb[kk * (size_t)N + nn]);
  }
}

// ------------------------------------------------ GEMM + bias + ELU -> bf16
// A: bf16 [M,K] row-major.  Bt: bf16 [N,K] row-major (transposed weights).
// Block = 256 threads = 8 waves.  Block tile: (MT*16) rows x (8 waves * NT*16)
// cols.  Inner loop is register double-buffered (ping-pong) so global loads
// for k-step ks+1 are in flight while WMMAs consume k-step ks.
template <int MT, int NT>
__global__ __launch_bounds__(256)
void gemm_bias_elu_bf16(const unsigned short* __restrict__ A,
                        const unsigned short* __restrict__ Bt,
                        const float* __restrict__ bias,
                        unsigned short* __restrict__ C,
                        int M, int N, int K) {
  const int lane  = threadIdx.x & 31;
  const int wave  = threadIdx.x >> 5;
  const int khalf = lane >> 4;
  const int lrow  = lane & 15;
  const int mbase = blockIdx.x * (MT * 16);
  const int nbase = (blockIdx.y * 8 + wave) * (NT * 16);
  const int ksteps = K >> 5;

  const unsigned short* arow[MT];
#pragma unroll
  for (int i = 0; i < MT; ++i)
    arow[i] = A + (size_t)(mbase + i * 16 + lrow) * K;
  const unsigned short* brow[NT];
  int ncol[NT];
#pragma unroll
  for (int j = 0; j < NT; ++j) {
    ncol[j] = nbase + j * 16 + lrow;
    brow[j] = Bt + (size_t)ncol[j] * K;
  }

  const v8f vzero = {0.f, 0.f, 0.f, 0.f, 0.f, 0.f, 0.f, 0.f};
  v8f acc[MT][NT];
#pragma unroll
  for (int i = 0; i < MT; ++i)
#pragma unroll
    for (int j = 0; j < NT; ++j) acc[i][j] = vzero;

  v16bf a0[MT], a1[MT], b0[NT], b1[NT];
#pragma unroll
  for (int i = 0; i < MT; ++i) a0[i] = load_frag_g(arow[i], 0, khalf);
#pragma unroll
  for (int j = 0; j < NT; ++j) b0[j] = load_frag_g(brow[j], 0, khalf);

#pragma unroll 1
  for (int ks = 0; ks < ksteps; ks += 2) {
    const int k1 = (ks + 1) * 32;
#pragma unroll
    for (int i = 0; i < MT; ++i) a1[i] = load_frag_g(arow[i], k1, khalf);
#pragma unroll
    for (int j = 0; j < NT; ++j) b1[j] = load_frag_g(brow[j], k1, khalf);
#pragma unroll
    for (int i = 0; i < MT; ++i)
#pragma unroll
      for (int j = 0; j < NT; ++j) acc[i][j] = wmma_bf16(a0[i], b0[j], acc[i][j]);

    const int k2 = (ks + 2 < ksteps) ? (ks + 2) * 32 : 0;  // dummy reload on last iter
#pragma unroll
    for (int i = 0; i < MT; ++i) a0[i] = load_frag_g(arow[i], k2, khalf);
#pragma unroll
    for (int j = 0; j < NT; ++j) b0[j] = load_frag_g(brow[j], k2, khalf);
#pragma unroll
    for (int i = 0; i < MT; ++i)
#pragma unroll
      for (int j = 0; j < NT; ++j) acc[i][j] = wmma_bf16(a1[i], b1[j], acc[i][j]);
  }

#pragma unroll
  for (int i = 0; i < MT; ++i)
#pragma unroll
    for (int j = 0; j < NT; ++j) {
      const float bv = bias[ncol[j]];
      unsigned short* cp =
          C + (size_t)(mbase + i * 16 + khalf * 8) * N + ncol[j];
#pragma unroll
      for (int r = 0; r < 8; ++r) {
        float v = acc[i][j][r] + bv;
        cp[(size_t)r * N] = f2bf(elu(v));
      }
    }
}

// ------------------------------------------------------ grouped head kernel
// blockIdx.y = intention n, blockIdx.x = 16-row tile within that bucket.
// Layer1: gathered z-rows x Wh1t[n] (+bias, ELU) -> LDS bf16 tile [16,1024].
// Layer2: LDS tile x Wh2t[n] (+bias) -> logits scattered to d_out rows.
#define TSH_STRIDE 1032  // 1024 + 8 bf16 pad -> rotates LDS banks per row
__global__ __launch_bounds__(256)
void head_kernel(const unsigned short* __restrict__ zb,    // [BATCH, HEAD_IN]
                 const unsigned short* __restrict__ Wh1t,  // [16, HEAD_HIDDEN, HEAD_IN]
                 const float* __restrict__ bh1,            // [16, HEAD_HIDDEN]
                 const unsigned short* __restrict__ Wh2t,  // [16, HEAD_OUT, HEAD_HIDDEN]
                 const float* __restrict__ bh2,            // [16, HEAD_OUT]
                 const int* __restrict__ counts,
                 const int* __restrict__ idx,
                 float* __restrict__ out) {                // [BATCH, HEAD_OUT] (logits here)
  __shared__ unsigned short t_sh[16 * TSH_STRIDE];
  __shared__ int rows[16];

  const int n    = blockIdx.y;
  const int cnt  = counts[n];
  const int tile = blockIdx.x;
  if (tile * 16 >= cnt) return;  // uniform across block

  if (threadIdx.x < 16) {
    int s = tile * 16 + (int)threadIdx.x;
    if (s > cnt - 1) s = cnt - 1;  // duplicate a valid row; result discarded
    rows[threadIdx.x] = idx[n * BATCH + s];
  }
  __syncthreads();

  const int lane  = threadIdx.x & 31;
  const int wave  = threadIdx.x >> 5;
  const int khalf = lane >> 4;
  const int lrow  = lane & 15;
  const v8f vzero = {0.f, 0.f, 0.f, 0.f, 0.f, 0.f, 0.f, 0.f};

  const unsigned short* zrow = zb + (size_t)rows[lrow] * HEAD_IN;
  const unsigned short* W1n  = Wh1t + (size_t)n * (HEAD_HIDDEN * HEAD_IN);
  const float*          b1n  = bh1 + n * HEAD_HIDDEN;

  // ---- layer 1: wave covers columns [wave*128, wave*128+128)
  for (int nt0 = 0; nt0 < 8; nt0 += 4) {
    v8f acc[4];
    const unsigned short* brow[4];
    int col[4];
#pragma unroll
    for (int j = 0; j < 4; ++j) {
      col[j]  = wave * 128 + (nt0 + j) * 16 + lrow;
      brow[j] = W1n + (size_t)col[j] * HEAD_IN;
      acc[j]  = vzero;
    }
    v16bf a0, a1, b0[4], b1[4];
    a0 = load_frag_g(zrow, 0, khalf);
#pragma unroll
    for (int j = 0; j < 4; ++j) b0[j] = load_frag_g(brow[j], 0, khalf);
#pragma unroll 1
    for (int ks = 0; ks < 32; ks += 2) {
      const int k1 = (ks + 1) * 32;
      a1 = load_frag_g(zrow, k1, khalf);
#pragma unroll
      for (int j = 0; j < 4; ++j) b1[j] = load_frag_g(brow[j], k1, khalf);
#pragma unroll
      for (int j = 0; j < 4; ++j) acc[j] = wmma_bf16(a0, b0[j], acc[j]);
      const int k2 = (ks + 2 < 32) ? (ks + 2) * 32 : 0;
      a0 = load_frag_g(zrow, k2, khalf);
#pragma unroll
      for (int j = 0; j < 4; ++j) b0[j] = load_frag_g(brow[j], k2, khalf);
#pragma unroll
      for (int j = 0; j < 4; ++j) acc[j] = wmma_bf16(a1, b1[j], acc[j]);
    }
#pragma unroll
    for (int j = 0; j < 4; ++j) {
      const float bv = b1n[col[j]];
#pragma unroll
      for (int r = 0; r < 8; ++r) {
        float v = acc[j][r] + bv;
        t_sh[(r + khalf * 8) * TSH_STRIDE + col[j]] = f2bf(elu(v));
      }
    }
  }
  __syncthreads();

  // ---- layer 2: wave covers output columns [wave*64, wave*64+64)
  const unsigned short* W2n = Wh2t + (size_t)n * (HEAD_OUT * HEAD_HIDDEN);
  const float*          b2n = bh2 + n * HEAD_OUT;
  {
    auto load_lds_frag = [&](int k0) -> v16bf {
      FragAB f;
      const unsigned short* p = &t_sh[lrow * TSH_STRIDE + k0 + khalf * 8];
      f.q[0] = *reinterpret_cast<const v4u*>(p);
      f.q[1] = *reinterpret_cast<const v4u*>(p + 16);
      return f.v;
    };
    v8f acc[4];
    const unsigned short* brow[4];
    int col[4];
#pragma unroll
    for (int j = 0; j < 4; ++j) {
      col[j]  = wave * 64 + j * 16 + lrow;
      brow[j] = W2n + (size_t)col[j] * HEAD_HIDDEN;
      acc[j]  = vzero;
    }
    v16bf a0, a1, b0[4], b1[4];
    a0 = load_lds_frag(0);
#pragma unroll
    for (int j = 0; j < 4; ++j) b0[j] = load_frag_g(brow[j], 0, khalf);
#pragma unroll 1
    for (int ks = 0; ks < 32; ks += 2) {
      const int k1 = (ks + 1) * 32;
      a1 = load_lds_frag(k1);
#pragma unroll
      for (int j = 0; j < 4; ++j) b1[j] = load_frag_g(brow[j], k1, khalf);
#pragma unroll
      for (int j = 0; j < 4; ++j) acc[j] = wmma_bf16(a0, b0[j], acc[j]);
      const int k2 = (ks + 2 < 32) ? (ks + 2) * 32 : 0;
      a0 = load_lds_frag(k2);
#pragma unroll
      for (int j = 0; j < 4; ++j) b0[j] = load_frag_g(brow[j], k2, khalf);
#pragma unroll
      for (int j = 0; j < 4; ++j) acc[j] = wmma_bf16(a1, b1[j], acc[j]);
    }
#pragma unroll
    for (int j = 0; j < 4; ++j) {
      const float bv = b2n[col[j]];
#pragma unroll
      for (int r = 0; r < 8; ++r) {
        const int m = r + khalf * 8;
        const int s = tile * 16 + m;
        if (s < cnt)
          out[(size_t)rows[m] * HEAD_OUT + col[j]] = acc[j][r] + bv;
      }
    }
  }
}

// ------------------------------------------- softmax over each row of d_out
__global__ __launch_bounds__(32)
void softmax_rows(float* __restrict__ out) {
  float* p = out + (size_t)blockIdx.x * HEAD_OUT;
  const int lane = threadIdx.x;
  float v[16];
  float mx = -1e30f;
#pragma unroll
  for (int i = 0; i < 16; ++i) { v[i] = p[lane + i * 32]; mx = fmaxf(mx, v[i]); }
#pragma unroll
  for (int o = 16; o >= 1; o >>= 1) mx = fmaxf(mx, __shfl_xor(mx, o, 32));
  float sum = 0.f;
#pragma unroll
  for (int i = 0; i < 16; ++i) { v[i] = __expf(v[i] - mx); sum += v[i]; }
#pragma unroll
  for (int o = 16; o >= 1; o >>= 1) sum += __shfl_xor(sum, o, 32);
  const float inv = 1.0f / sum;
#pragma unroll
  for (int i = 0; i < 16; ++i) p[lane + i * 32] = v[i] * inv;
}

// ---------------------------------------------------------------- launcher
extern "C" void kernel_launch(void* const* d_in, const int* in_sizes, int n_in,
                              void* d_out, int out_size, void* d_ws, size_t ws_size,
                              hipStream_t stream) {
  const float* x      = (const float*)d_in[0];
  const int*   intent = (const int*)d_in[1];
  const float* W1     = (const float*)d_in[2];
  const float* b1     = (const float*)d_in[3];
  const float* W2     = (const float*)d_in[4];
  const float* b2     = (const float*)d_in[5];
  const float* Wh1    = (const float*)d_in[6];
  const float* bh1    = (const float*)d_in[7];
  const float* Wh2    = (const float*)d_in[8];
  const float* bh2    = (const float*)d_in[9];
  float* out = (float*)d_out;

  // workspace carve-up (~117 MB total)
  size_t off = 0;
  auto take = [&](size_t bytes) -> void* {
    void* p = (char*)d_ws + off;
    off += (bytes + 255) & ~(size_t)255;
    return p;
  };
  int*            counts = (int*)take(NUM_INT * sizeof(int));
  int*            idx    = (int*)take((size_t)NUM_INT * BATCH * sizeof(int));
  unsigned short* xb     = (unsigned short*)take((size_t)BATCH * STATE_DIM * 2);
  unsigned short* W1t    = (unsigned short*)take((size_t)BASE_HIDDEN * STATE_DIM * 2);
  unsigned short* W2t    = (unsigned short*)take((size_t)HEAD_IN * BASE_HIDDEN * 2);
  unsigned short* Wh1t   = (unsigned short*)take((size_t)NUM_INT * HEAD_HIDDEN * HEAD_IN * 2);
  unsigned short* Wh2t   = (unsigned short*)take((size_t)NUM_INT * HEAD_OUT * HEAD_HIDDEN * 2);
  unsigned short* hb     = (unsigned short*)take((size_t)BATCH * BASE_HIDDEN * 2);
  unsigned short* zb     = (unsigned short*)take((size_t)BATCH * HEAD_IN * 2);

  // 1) bucket samples by intention
  bucket_init<<<1, 32, 0, stream>>>(counts);
  bucket<<<BATCH / 256, 256, 0, stream>>>(intent, counts, idx);

  // 2) convert activations / transpose-convert weights to bf16 [N,K]
  cvt_bf16<<<1024, 256, 0, stream>>>(x, xb, (size_t)BATCH * STATE_DIM);
  transpose_cvt_bf16<<<dim3(512, 1), 256, 0, stream>>>(W1, W1t, STATE_DIM, BASE_HIDDEN);
  transpose_cvt_bf16<<<dim3(512, 1), 256, 0, stream>>>(W2, W2t, BASE_HIDDEN, HEAD_IN);
  transpose_cvt_bf16<<<dim3(512, NUM_INT), 256, 0, stream>>>(Wh1, Wh1t, HEAD_IN, HEAD_HIDDEN);
  transpose_cvt_bf16<<<dim3(512, NUM_INT), 256, 0, stream>>>(Wh2, Wh2t, HEAD_HIDDEN, HEAD_OUT);

  // 3) trunk: 2x4 register tiling, 8 waves/block
  // block covers 32 rows x 512 cols
  gemm_bias_elu_bf16<2, 4><<<dim3(BATCH / 32, BASE_HIDDEN / 512), 256, 0, stream>>>(
      xb, W1t, b1, hb, BATCH, BASE_HIDDEN, STATE_DIM);
  gemm_bias_elu_bf16<2, 4><<<dim3(BATCH / 32, HEAD_IN / 512), 256, 0, stream>>>(
      hb, W2t, b2, zb, BATCH, HEAD_IN, BASE_HIDDEN);

  // 4) grouped heads (only the selected head per sample) -> logits in d_out
  head_kernel<<<dim3(BATCH / 16, NUM_INT), 256, 0, stream>>>(
      zb, Wh1t, bh1, Wh2t, bh2, counts, idx, out);

  // 5) softmax over each output row
  softmax_rows<<<BATCH, 32, 0, stream>>>(out);
}
